// RNN_54400055771283
// MI455X (gfx1250) — matrix operand order repaired and checked by reading, not compiled, not run on hardware
//
#include <hip/hip_runtime.h>
#include <hip/hip_bf16.h>
#include <math.h>

#define EE 512
#define HH 512
#define BB 64
#define SS 512
#define BS (BB*SS)
#define LROW 520           // LDS row stride in bf16 elems (1040B: breaks 256B bank period)
#define HBUFE (16 * LROW)  // one h buffer, elems

typedef __attribute__((ext_vector_type(16))) __bf16 v16bf;
typedef __attribute__((ext_vector_type(8)))  __bf16 v8bf;
typedef __attribute__((ext_vector_type(8)))  float  v8f;

// ---------------------------------------------------------------- utilities
__global__ void cvt_bf16(const float* __restrict__ src, __bf16* __restrict__ dst, int n) {
    int i = blockIdx.x * blockDim.x + threadIdx.x;
    if (i < n) dst[i] = (__bf16)src[i];
}

// ------------------------------------------------- GEMM 1: gather + project
// out[i,n] = sum_e emb[tokens[i],e]*W[n,e] + bias1[n] + bias2[n]
__global__ __launch_bounds__(256)
void gemm_embed(const int* __restrict__ tokens, const float* __restrict__ emb,
                const __bf16* __restrict__ Wb, const float* __restrict__ bias1,
                const float* __restrict__ bias2, float* __restrict__ out) {
    const int lane  = threadIdx.x & 31;
    const int wave  = threadIdx.x >> 5;
    const int tile  = blockIdx.x * 8 + wave;
    const int nTile = tile & 31;          // H/16 = 32 tiles in N
    const int mTile = tile >> 5;
    const int m0 = mTile * 16, n0 = nTile * 16;
    const int ln = lane & 15, lh = lane >> 4;   // lh = 0/1 half selector
    const int kh = lh * 16;

    const int    tok  = tokens[m0 + ln];
    const float* arow = emb + (size_t)tok * EE;                 // A row, M = lane%16
    const __bf16* brow = Wb + (size_t)(n0 + ln) * EE + lh * 16; // B column strip

    v8f acc = {};
    #pragma unroll
    for (int kk = 0; kk < EE; kk += 32) {
        v16bf a;
        #pragma unroll
        for (int j = 0; j < 8; ++j) a[j]     = (__bf16)arow[kk + kh + j];
        #pragma unroll
        for (int j = 0; j < 8; ++j) a[j + 8] = (__bf16)arow[kk + kh + 8 + j];
        v16bf b = *(const v16bf*)(brow + kk);
        acc = __builtin_amdgcn_wmma_f32_16x16x32_bf16(false, a, false, b,
                                                      (short)0, acc, false, false);
    }
    const int   n  = n0 + ln;
    const float bs = bias1[n] + bias2[n];
    #pragma unroll
    for (int j = 0; j < 8; ++j) {
        int r = m0 + j + lh * 8;
        out[(size_t)r * HH + n] = acc[j] + bs;
    }
}

// ------------------------------------------------- GEMM 2: bf16 A (layer 1)
__global__ __launch_bounds__(256)
void gemm_bf16(const __bf16* __restrict__ A, const __bf16* __restrict__ Wb,
               const float* __restrict__ bias1, const float* __restrict__ bias2,
               float* __restrict__ out) {
    const int lane  = threadIdx.x & 31;
    const int wave  = threadIdx.x >> 5;
    const int tile  = blockIdx.x * 8 + wave;
    const int nTile = tile & 31;
    const int mTile = tile >> 5;
    const int m0 = mTile * 16, n0 = nTile * 16;
    const int ln = lane & 15, lh = lane >> 4;
    const int kh = lh * 16;

    const __bf16* arow = A  + (size_t)(m0 + ln) * HH;
    const __bf16* brow = Wb + (size_t)(n0 + ln) * HH + lh * 16;

    v8f acc = {};
    #pragma unroll
    for (int kk = 0; kk < HH; kk += 32) {
        v8bf c0 = *(const v8bf*)(arow + kk + kh);
        v8bf c1 = *(const v8bf*)(arow + kk + kh + 8);
        v16bf a;
        #pragma unroll
        for (int j = 0; j < 8; ++j) { a[j] = c0[j]; a[j + 8] = c1[j]; }
        v16bf b = *(const v16bf*)(brow + kk);
        acc = __builtin_amdgcn_wmma_f32_16x16x32_bf16(false, a, false, b,
                                                      (short)0, acc, false, false);
    }
    const int   n  = n0 + ln;
    const float bs = bias1[n] + bias2[n];
    #pragma unroll
    for (int j = 0; j < 8; ++j) {
        int r = m0 + j + lh * 8;
        out[(size_t)r * HH + n] = acc[j] + bs;
    }
}

// ------------------------------------------------- recurrent scan (1 layer)
// 4 workgroups (16 batch rows each), 16 waves; wave owns 2 N-tiles.
// Double-buffered bf16 h in LDS -> ONE barrier per step on the serial chain.
// Split-K accumulators halve the WMMA dependency depth per step.
__global__ __launch_bounds__(512)
void rnn_scan(const float* __restrict__ xw, const __bf16* __restrict__ Whh,
              __bf16* __restrict__ hout) {
    __shared__ __bf16 hls[2 * HBUFE];
    const int lane = threadIdx.x & 31;
    const int wave = threadIdx.x >> 5;          // 0..15
    const int b0   = blockIdx.x * 16;           // batch rows b0..b0+15
    const int ln = lane & 15, lh = lane >> 4;
    const int kh = lh * 16;
    const int n0a = (2 * wave) * 16, n0b = n0a + 16;

    for (int i = threadIdx.x; i < HBUFE; i += blockDim.x) hls[i] = (__bf16)0.f;
    __syncthreads();

    const __bf16* browA = Whh + (size_t)(n0a + ln) * HH + lh * 16;
    const __bf16* browB = Whh + (size_t)(n0b + ln) * HH + lh * 16;

    for (int t = 0; t < SS; ++t) {
        const int rd = t & 1, wr = rd ^ 1;
        const __bf16* hrow = hls + rd * HBUFE + ln * LROW;  // A row, M = lane%16
        __bf16*       hnew = hls + wr * HBUFE;

        v8f acc0, acc1;                         // chain A: init with input projection
        v8f acc0b = {}, acc1b = {};             // chain B: second 32 k of each 64-k pair
        #pragma unroll
        for (int j = 0; j < 8; ++j) {
            size_t base = ((size_t)(b0 + j + lh * 8) * SS + t) * HH;
            acc0[j] = xw[base + n0a + ln];
            acc1[j] = xw[base + n0b + ln];
        }
        #pragma unroll
        for (int kk = 0; kk < HH; kk += 64) {
            // ---- first 32 of the pair -> chain A
            {
                v8bf c0 = *(const v8bf*)(hrow + kk + kh);
                v8bf c1 = *(const v8bf*)(hrow + kk + kh + 8);
                v16bf a;
                #pragma unroll
                for (int j = 0; j < 8; ++j) { a[j] = c0[j]; a[j + 8] = c1[j]; }
                v16bf bA = *(const v16bf*)(browA + kk);
                v16bf bB = *(const v16bf*)(browB + kk);
                acc0 = __builtin_amdgcn_wmma_f32_16x16x32_bf16(false, a, false, bA,
                                                               (short)0, acc0, false, false);
                acc1 = __builtin_amdgcn_wmma_f32_16x16x32_bf16(false, a, false, bB,
                                                               (short)0, acc1, false, false);
            }
            // ---- second 32 of the pair -> chain B (independent accumulator)
            {
                v8bf c0 = *(const v8bf*)(hrow + kk + 32 + kh);
                v8bf c1 = *(const v8bf*)(hrow + kk + 32 + kh + 8);
                v16bf a;
                #pragma unroll
                for (int j = 0; j < 8; ++j) { a[j] = c0[j]; a[j + 8] = c1[j]; }
                v16bf bA = *(const v16bf*)(browA + kk + 32);
                v16bf bB = *(const v16bf*)(browB + kk + 32);
                acc0b = __builtin_amdgcn_wmma_f32_16x16x32_bf16(false, a, false, bA,
                                                                (short)0, acc0b, false, false);
                acc1b = __builtin_amdgcn_wmma_f32_16x16x32_bf16(false, a, false, bB,
                                                                (short)0, acc1b, false, false);
            }
        }
        #pragma unroll
        for (int j = 0; j < 8; ++j) {
            acc0[j] = tanhf(acc0[j] + acc0b[j]);
            acc1[j] = tanhf(acc1[j] + acc1b[j]);
        }

        // write h_t into the OTHER buffer: no conflict with this step's reads
        #pragma unroll
        for (int j = 0; j < 8; ++j) {
            int r = j + lh * 8;
            __bf16 ha = (__bf16)acc0[j];
            __bf16 hb = (__bf16)acc1[j];
            hnew[r * LROW + n0a + ln] = ha;
            hnew[r * LROW + n0b + ln] = hb;
            size_t base = ((size_t)(b0 + r) * SS + t) * HH;
            hout[base + n0a + ln] = ha;
            hout[base + n0b + ln] = hb;
        }
        __syncthreads();   // single barrier per step: h_t visible for step t+1
    }
}

// ------------------------------------------------- final FC (C=2, tiny)
__global__ void fc_kernel(const __bf16* __restrict__ h, const float* __restrict__ Wfc,
                          const float* __restrict__ bfc, float* __restrict__ out) {
    int i = blockIdx.x * blockDim.x + threadIdx.x;
    if (i >= BS) return;
    const __bf16* hr = h + (size_t)i * HH;
    float a0 = 0.f, a1 = 0.f;
    for (int k = 0; k < HH; ++k) {
        float x = (float)hr[k];
        a0 += x * Wfc[k];
        a1 += x * Wfc[HH + k];
    }
    out[i * 2 + 0] = a0 + bfc[0];
    out[i * 2 + 1] = a1 + bfc[1];
}

// ---------------------------------------------------------------- launcher
extern "C" void kernel_launch(void* const* d_in, const int* in_sizes, int n_in,
                              void* d_out, int out_size, void* d_ws, size_t ws_size,
                              hipStream_t stream) {
    const int*   tokens = (const int*)  d_in[0];
    const float* emb    = (const float*)d_in[1];
    const float* W_ih0  = (const float*)d_in[2];
    const float* b_ih0  = (const float*)d_in[3];
    const float* W_hh0  = (const float*)d_in[4];
    const float* b_hh0  = (const float*)d_in[5];
    const float* W_ih1  = (const float*)d_in[6];
    const float* b_ih1  = (const float*)d_in[7];
    const float* W_hh1  = (const float*)d_in[8];
    const float* b_hh1  = (const float*)d_in[9];
    const float* W_fc   = (const float*)d_in[10];
    const float* b_fc   = (const float*)d_in[11];

    char* ws = (char*)d_ws;
    const size_t WSZ = (size_t)HH * HH * sizeof(__bf16);      // 512 KB
    __bf16* wih0b = (__bf16*)(ws + 0 * WSZ);
    __bf16* whh0b = (__bf16*)(ws + 1 * WSZ);
    __bf16* wih1b = (__bf16*)(ws + 2 * WSZ);
    __bf16* whh1b = (__bf16*)(ws + 3 * WSZ);
    float*  xwbuf = (float*) (ws + 4 * WSZ);                  // 64 MB f32
    __bf16* hbuf  = (__bf16*)(ws + 4 * WSZ + (size_t)BS * HH * sizeof(float)); // 32 MB

    const int wn = HH * HH;
    cvt_bf16<<<(wn + 255) / 256, 256, 0, stream>>>(W_ih0, wih0b, wn);
    cvt_bf16<<<(wn + 255) / 256, 256, 0, stream>>>(W_hh0, whh0b, wn);
    cvt_bf16<<<(wn + 255) / 256, 256, 0, stream>>>(W_ih1, wih1b, wn);
    cvt_bf16<<<(wn + 255) / 256, 256, 0, stream>>>(W_hh1, whh1b, wn);

    const int tiles  = (BS / 16) * (HH / 16);  // 2048 * 32 = 65536 tiles
    const int blocks = tiles / 8;              // 8 waves per 256-thread block

    // layer 0: input projection (+both biases), then scan
    gemm_embed<<<blocks, 256, 0, stream>>>(tokens, emb, wih0b, b_ih0, b_hh0, xwbuf);
    rnn_scan  <<<BB / 16, 512, 0, stream>>>(xwbuf, whh0b, hbuf);

    // layer 1: input projection from h0 (bf16), then scan (reuses buffers)
    gemm_bf16 <<<blocks, 256, 0, stream>>>(hbuf, wih1b, b_ih1, b_hh1, xwbuf);
    rnn_scan  <<<BB / 16, 512, 0, stream>>>(xwbuf, whh1b, hbuf);

    // classifier head
    fc_kernel <<<(BS + 255) / 256, 256, 0, stream>>>(hbuf, W_fc, b_fc, (float*)d_out);
}
